// ScannedRNN_11630771438401
// MI455X (gfx1250) — matrix operand order repaired and checked by reading, not compiled, not run on hardware
//
#include <hip/hip_runtime.h>

// Problem constants (reference: T,B,D,H = 512,256,128,128)
#define T_DIM 512
#define B_DIM 256
#define D_DIM 128
#define H_DIM 128
#define N4H   512           // 4*H gate width
#define LDS_HS 136          // padded f16 row stride (breaks LDS bank conflicts)

typedef __attribute__((ext_vector_type(16))) _Float16 v16h;
typedef __attribute__((ext_vector_type(8)))  float    v8f;

union FragU { v16h v; uint4 q[2]; };

// ---- WMMA fragment loaders -------------------------------------------------
// B matrix 32x16 (KxN), f16, from [n][k] transposed weight image (k contiguous):
// lanes 0-15 hold K=k0..k0+15 of column n0+lane; lanes 16-31 hold K=k0+16..k0+31.
__device__ __forceinline__ v16h load_b_frag(const _Float16* __restrict__ matT,
                                            int n0, int k0, int lane) {
  int n = n0 + (lane & 15);
  int k = k0 + ((lane & 16) ? 16 : 0);
  FragU u;
  const uint4* p = (const uint4*)(matT + (size_t)n * 128 + k);
  u.q[0] = p[0];
  u.q[1] = p[1];
  return u.v;
}

// A matrix 16x32 (MxK), f16, from LDS tile with padded stride LDS_HS.
// lane<16: row=lane, K = k0+{0..15}; lane>=16: row=lane-16, K = k0+{16..31}.
__device__ __forceinline__ v16h load_a_frag_lds(const _Float16* hbuf, int k0, int lane) {
  int m   = lane & 15;
  int klo = k0 + ((lane & 16) ? 16 : 0);
  FragU u;
  u.q[0] = *(const uint4*)(hbuf + m * LDS_HS + klo);
  u.q[1] = *(const uint4*)(hbuf + m * LDS_HS + klo + 8);
  return u.v;
}

__device__ __forceinline__ float fast_sigmoid(float x) {
  return __builtin_amdgcn_rcpf(1.0f + __expf(-x));
}
__device__ __forceinline__ float fast_tanh(float x) {
  float e = __expf(-2.0f * x);
  return (1.0f - e) * __builtin_amdgcn_rcpf(1.0f + e);
}

// Async DMA: copy four 16-byte chunks, 4 KB apart, global -> LDS, tracked by ASYNCcnt.
// IOFFSET applies to BOTH lds and global addresses (ISA 08 §4.4), so chunked
// copies only need immediate offsets.
__device__ __forceinline__ void async_copy_b128(unsigned int lds_addr, const void* gaddr) {
  asm volatile("global_load_async_to_lds_b128 %0, %1, off"
               :: "v"(lds_addr), "v"(gaddr) : "memory");
  asm volatile("global_load_async_to_lds_b128 %0, %1, off offset:4096"
               :: "v"(lds_addr), "v"(gaddr) : "memory");
  asm volatile("global_load_async_to_lds_b128 %0, %1, off offset:8192"
               :: "v"(lds_addr), "v"(gaddr) : "memory");
  asm volatile("global_load_async_to_lds_b128 %0, %1, off offset:12288"
               :: "v"(lds_addr), "v"(gaddr) : "memory");
}
__device__ __forceinline__ void wait_async0() {
  asm volatile("s_wait_asynccnt 0x0" ::: "memory");
}

// ---- Kernel 1: transpose+convert Wi/Wh (f32 [k][n]) -> f16 [n][k] ----------
__global__ void prep_weights(const float* __restrict__ Wi, const float* __restrict__ Wh,
                             _Float16* __restrict__ wiT, _Float16* __restrict__ whT) {
  int idx = blockIdx.x * blockDim.x + threadIdx.x;   // 0 .. 2*65536-1
  int m = idx >> 16;
  int e = idx & 65535;
  int n = e >> 7;          // 0..511
  int k = e & 127;         // 0..127
  const float* src = m ? Wh : Wi;
  _Float16*    dst = m ? whT : wiT;
  dst[(size_t)n * 128 + k] = (_Float16)src[(size_t)k * N4H + n];
}

// ---- Kernel 2 (precompute path): Zpre = x @ Wi + b, f16 out ----------------
// One block = 16 rows of (T*B, 128); 8 waves, each wave = 64 output columns.
// All 16 Wi B-fragments preloaded into registers (overlaps the LDS tile fill),
// so the K loop is pure ds_load + wmma.
__global__ void __launch_bounds__(256) gemm_xwi(const float* __restrict__ x,
                                                const _Float16* __restrict__ wiT,
                                                const float* __restrict__ bias,
                                                _Float16* __restrict__ zpre) {
  __shared__ __align__(16) _Float16 lds_x[16 * LDS_HS];
  const int tid = threadIdx.x, lane = tid & 31, wave = tid >> 5;
  const size_t r0 = (size_t)blockIdx.x * 16;
  const int n0 = wave * 64;

  // Kick off register preload of Wi fragments first (L2-hot, shared by all blocks).
  v16h bf[4][4];
  #pragma unroll
  for (int ks = 0; ks < 4; ++ks)
    #pragma unroll
    for (int i = 0; i < 4; ++i)
      bf[ks][i] = load_b_frag(wiT, n0 + 16 * i, ks * 32, lane);

  #pragma unroll
  for (int j = 0; j < 8; ++j) {
    int idx = tid + j * 256;                 // 0..2047
    int bb = idx >> 7, hc = idx & 127;
    lds_x[bb * LDS_HS + hc] = (_Float16)x[(r0 + bb) * D_DIM + hc];
  }
  __syncthreads();

  v8f acc[4];
  #pragma unroll
  for (int i = 0; i < 4; ++i) {
    float bv = bias[n0 + 16 * i + (lane & 15)];
    #pragma unroll
    for (int r = 0; r < 8; ++r) acc[i][r] = bv;
  }
  #pragma unroll
  for (int ks = 0; ks < 4; ++ks) {
    v16h a = load_a_frag_lds(lds_x, ks * 32, lane);
    #pragma unroll
    for (int i = 0; i < 4; ++i)
      acc[i] = __builtin_amdgcn_wmma_f32_16x16x32_f16(false, a, false, bf[ks][i],
                                                      (short)0, acc[i], false, false);
  }
  const int mb = (lane & 16) ? 8 : 0;
  #pragma unroll
  for (int i = 0; i < 4; ++i) {
    int col = n0 + 16 * i + (lane & 15);
    #pragma unroll
    for (int r = 0; r < 8; ++r)
      zpre[(r0 + mb + r) * N4H + col] = (_Float16)acc[i][r];
  }
}

// ---- Kernel 3: persistent sequential scan ----------------------------------
// 16 blocks x 256 threads; block owns 16 batch rows for all T steps.
// Wave w owns gate columns [w*64, w*64+64); its 16 Wh B-fragments (128 VGPRs)
// stay register-resident across the entire scan. In the precompute path the
// step's Zpre tile (16 KB, contiguous) is DMA'd into LDS with
// GLOBAL_LOAD_ASYNC_TO_LDS_B128, overlapping the recurrent WMMA work.
template <bool FUSED>
__global__ void __launch_bounds__(256, 1) lstm_scan(
    const float* __restrict__ x, const unsigned char* __restrict__ resets,
    const float* __restrict__ c0, const float* __restrict__ h0,
    const _Float16* __restrict__ wiT, const _Float16* __restrict__ whT,
    const float* __restrict__ bias, const _Float16* __restrict__ zpre,
    float* __restrict__ ys) {
  __shared__ __align__(16) _Float16 lds_h[16 * LDS_HS];
  __shared__ __align__(16) _Float16 lds_x[FUSED ? 16 * LDS_HS : 8];
  __shared__ __align__(16) _Float16 zbuf[FUSED ? 8 : 16 * N4H];   // async staging
  __shared__ float lds_c[16 * 128];
  __shared__ float lds_z[16 * N4H];

  const int tid = threadIdx.x, lane = tid & 31, wave = tid >> 5;
  const int b0 = blockIdx.x * 16;
  const int n0 = wave * 64;
  // 32-bit LDS byte address of this thread's staging slot (flat-aperture low bits).
  const unsigned int zbuf_lds = (unsigned int)(uintptr_t)(&zbuf[0]) + tid * 16;

  // Prologue: load c0/h0 with resets[0] applied.
  #pragma unroll
  for (int j = 0; j < 8; ++j) {
    int idx = tid + j * 256;
    int bb = idx >> 7, hc = idx & 127;
    bool r = resets[(size_t)0 * B_DIM + b0 + bb] != 0;
    float cv = r ? 0.f : c0[(size_t)(b0 + bb) * H_DIM + hc];
    float hv = r ? 0.f : h0[(size_t)(b0 + bb) * H_DIM + hc];
    lds_c[idx] = cv;
    lds_h[bb * LDS_HS + hc] = (_Float16)hv;
  }

  // Register-resident Wh fragments: [kstep][coltile] = 16 x v16h = 128 VGPRs.
  v16h whf[4][4];
  #pragma unroll
  for (int ks = 0; ks < 4; ++ks)
    #pragma unroll
    for (int i = 0; i < 4; ++i)
      whf[ks][i] = load_b_frag(whT, n0 + 16 * i, ks * 32, lane);
  __syncthreads();

  for (int t = 0; t < T_DIM; ++t) {
    if (!FUSED) {
      // DMA this step's Zpre tile into LDS; completes under the WMMA work.
      const char* g = (const char*)zpre +
                      ((size_t)t * B_DIM + b0) * N4H * sizeof(_Float16) + tid * 16;
      async_copy_b128(zbuf_lds, g);
    } else {
      #pragma unroll
      for (int j = 0; j < 8; ++j) {
        int idx = tid + j * 256;
        int bb = idx >> 7, hc = idx & 127;
        lds_x[bb * LDS_HS + hc] = (_Float16)x[((size_t)t * B_DIM + b0 + bb) * D_DIM + hc];
      }
      __syncthreads();
    }

    // z tile = h @ Wh (+ x @ Wi when fused); fp32 accumulate via WMMA f16.
    v8f acc[4];
    #pragma unroll
    for (int i = 0; i < 4; ++i) {
      float init = FUSED ? bias[n0 + 16 * i + (lane & 15)] : 0.f;
      #pragma unroll
      for (int r = 0; r < 8; ++r) acc[i][r] = init;
    }
    #pragma unroll
    for (int ks = 0; ks < 4; ++ks) {
      v16h a = load_a_frag_lds(lds_h, ks * 32, lane);
      #pragma unroll
      for (int i = 0; i < 4; ++i)
        acc[i] = __builtin_amdgcn_wmma_f32_16x16x32_f16(false, a, false, whf[ks][i],
                                                        (short)0, acc[i], false, false);
    }
    if (FUSED) {
      #pragma unroll
      for (int ks = 0; ks < 4; ++ks) {
        v16h a = load_a_frag_lds(lds_x, ks * 32, lane);
        #pragma unroll
        for (int i = 0; i < 4; ++i) {
          v16h b = load_b_frag(wiT, n0 + 16 * i, ks * 32, lane);
          acc[i] = __builtin_amdgcn_wmma_f32_16x16x32_f16(false, a, false, b,
                                                          (short)0, acc[i], false, false);
        }
      }
    }

    // Scatter accumulators into the shared z tile for cross-gate combine.
    const int mb = (lane & 16) ? 8 : 0;
    #pragma unroll
    for (int i = 0; i < 4; ++i) {
      int col = n0 + 16 * i + (lane & 15);
      #pragma unroll
      for (int r = 0; r < 8; ++r) lds_z[(mb + r) * N4H + col] = acc[i][r];
    }

    if (!FUSED) wait_async0();   // this wave's 4 DMA chunks have landed in LDS
    __syncthreads();             // publish lds_z stores + async LDS writes

    // Elementwise gates + state update; pre-apply resets[t+1] to stored state.
    #pragma unroll
    for (int j = 0; j < 8; ++j) {
      int idx = tid + j * 256;
      int bb = idx >> 7, hc = idx & 127;
      float zi = lds_z[bb * N4H + hc];
      float zf = lds_z[bb * N4H + 128 + hc];
      float zg = lds_z[bb * N4H + 256 + hc];
      float zo = lds_z[bb * N4H + 384 + hc];
      if (!FUSED) {
        zi += (float)zbuf[bb * N4H + hc];
        zf += (float)zbuf[bb * N4H + 128 + hc];
        zg += (float)zbuf[bb * N4H + 256 + hc];
        zo += (float)zbuf[bb * N4H + 384 + hc];
      }
      float ig = fast_sigmoid(zi);
      float fg = fast_sigmoid(zf);
      float gg = fast_tanh(zg);
      float og = fast_sigmoid(zo);
      float c  = lds_c[idx];
      float nc = fg * c + ig * gg;
      float nh = og * fast_tanh(nc);
      ys[((size_t)t * B_DIM + b0 + bb) * H_DIM + hc] = nh;
      bool rn = (t + 1 < T_DIM) && (resets[(size_t)(t + 1) * B_DIM + b0 + bb] != 0);
      lds_c[idx] = rn ? 0.f : nc;
      lds_h[bb * LDS_HS + hc] = (_Float16)(rn ? 0.f : nh);
    }
    __syncthreads();   // retire zbuf/lds_z reads before next step's DMA/stores
  }
}

// ---- Host launcher ----------------------------------------------------------
extern "C" void kernel_launch(void* const* d_in, const int* in_sizes, int n_in,
                              void* d_out, int out_size, void* d_ws, size_t ws_size,
                              hipStream_t stream) {
  (void)in_sizes; (void)n_in; (void)out_size;
  const float*         x      = (const float*)d_in[0];
  const unsigned char* resets = (const unsigned char*)d_in[1];  // jax bool = 1 byte
  const float*         c0     = (const float*)d_in[2];
  const float*         h0     = (const float*)d_in[3];
  const float*         Wi     = (const float*)d_in[4];
  const float*         Wh     = (const float*)d_in[5];
  const float*         bias   = (const float*)d_in[6];
  float*               ys     = (float*)d_out;

  char* ws = (char*)d_ws;
  _Float16* whT  = (_Float16*)(ws);
  _Float16* wiT  = (_Float16*)(ws + (128 << 10));
  _Float16* zpre = (_Float16*)(ws + (256 << 10));
  const size_t need = (size_t)(256 << 10) + (size_t)T_DIM * B_DIM * N4H * sizeof(_Float16);
  const bool fused = (ws_size < need);   // constant per process -> deterministic path

  prep_weights<<<512, 256, 0, stream>>>(Wi, Wh, wiT, whT);
  if (!fused) {
    gemm_xwi<<<(T_DIM * B_DIM) / 16, 256, 0, stream>>>(x, wiT, bias, zpre);
    lstm_scan<false><<<B_DIM / 16, 256, 0, stream>>>(x, resets, c0, h0, wiT, whT, bias, zpre, ys);
  } else {
    lstm_scan<true><<<B_DIM / 16, 256, 0, stream>>>(x, resets, c0, h0, wiT, whT, bias, zpre, ys);
  }
}
